// SemanticRelationModeler_70068096467240
// MI455X (gfx1250) — matrix operand order repaired and checked by reading, not compiled, not run on hardware
//
#include <hip/hip_runtime.h>
#include <math.h>

// ---------------------------------------------------------------------------
// MI455X / gfx1250 implementation, round 3.
//  - v_wmma_f32_16x16x32_f16 for: projection GEMMs, fused pair classifier
//    GEMM ([P x 512] x [512 x 16]), AND the pair strength mat-vec
//    ([P x 256] x [256 x 16], sw2 zero-padded) -- no scalar dot loop left.
//  - Projections stored f16 with biases pre-folded; A-fragments built with
//    packed f16 ops (v_pk_add_f16 / v_pk_max) from 16B half-vector loads
//    (global_load_b128).
//  - LDS tiles for the log_softmax/argmax/sigmoid epilogue.
// ---------------------------------------------------------------------------

typedef __attribute__((ext_vector_type(16))) _Float16 v16h;
typedef __attribute__((ext_vector_type(8)))  _Float16 v8h;
typedef __attribute__((ext_vector_type(8)))  float    v8f;

#define NN      512
#define DD      512
#define NPAIRS  130816   // 512*511/2, divisible by 16

__device__ __forceinline__ int tri_start(int i) {
    return (i * (2 * NN - 1 - i)) >> 1;   // product always even
}

__device__ __forceinline__ v8h ld8(const _Float16* p) {
    return *(const v8h*)p;                // 16B -> global_load_b128
}

__device__ __forceinline__ v8h relu8(v8h x) {
#if __has_builtin(__builtin_elementwise_max)
    v8h z = {};
    return __builtin_elementwise_max(x, z);   // v_pk_max
#else
#pragma unroll
    for (int e = 0; e < 8; ++e)
        x[e] = (x[e] > (_Float16)0) ? x[e] : (_Float16)0;
    return x;
#endif
}

__device__ __forceinline__ v16h cat8(v8h lo, v8h hi) {
    return __builtin_shufflevector(lo, hi, 0, 1, 2, 3, 4, 5, 6, 7,
                                           8, 9, 10, 11, 12, 13, 14, 15);
}

// ------------------------- conversion kernels ------------------------------

__global__ void cvt_f32_to_f16(const float* __restrict__ in,
                               _Float16* __restrict__ out, int n) {
    int idx = blockIdx.x * blockDim.x + threadIdx.x;
    if (idx < n) out[idx] = (_Float16)in[idx];
}

// out[n*K + k] = in[k*Ncols + n]   (store B transposed, K-contiguous per col)
__global__ void transpose_to_f16(const float* __restrict__ in,
                                 _Float16* __restrict__ out,
                                 int K, int Ncols) {
    int idx = blockIdx.x * blockDim.x + threadIdx.x;
    if (idx >= K * Ncols) return;
    int n = idx / K;
    int k = idx - n * K;
    out[n * K + k] = (_Float16)in[k * Ncols + n];
}

// cw2 [512 x 10] -> cw2t [16 x 512] f16, cols 10..15 zero-padded
__global__ void build_cw2t(const float* __restrict__ cw2,
                           _Float16* __restrict__ out) {
    int idx = blockIdx.x * blockDim.x + threadIdx.x;
    if (idx >= 16 * 512) return;
    int n = idx >> 9;
    int k = idx & 511;
    out[n * 512 + k] = (n < 10) ? (_Float16)cw2[k * 10 + n] : (_Float16)0.f;
}

// sw2 [256 x 1] -> sw2t [16 x 256] f16, cols 1..15 zero-padded
__global__ void build_sw2t(const float* __restrict__ sw2,
                           _Float16* __restrict__ out) {
    int idx = blockIdx.x * blockDim.x + threadIdx.x;
    if (idx >= 16 * 256) return;
    int n = idx >> 8;
    int k = idx & 255;
    out[n * 256 + k] = (n == 0) ? (_Float16)sw2[k] : (_Float16)0.f;
}

// ------------------------- projection GEMM ---------------------------------
// C_f16[M x Nc] = A_f16[M x K] * B (+ optional bias), B given transposed:
// column n of B is the K-contiguous row  BT + n*ldbt.
// One wave per 16x16 C tile; 4 waves per block.

__global__ __launch_bounds__(128)
void proj_gemm_f16(const _Float16* __restrict__ A,
                   const _Float16* __restrict__ BT, int ldbt,
                   _Float16* __restrict__ C,
                   const float* __restrict__ bias,   // may be nullptr
                   int M, int K, int Nc) {
    int lane = threadIdx.x & 31;
    int wave = threadIdx.x >> 5;
    int tile = blockIdx.x * 4 + wave;
    int tilesN = Nc >> 4;
    int tm = (tile / tilesN) << 4;
    int tn = (tile % tilesN) << 4;
    int nm = lane & 15;
    int hi = lane >> 4;
    int koffA = 8 * hi;      // A frag: regs0-3 K=8*hi+0..7, regs4-7 K=16+8*hi+0..7
    int koffB = 16 * hi;     // unused; B frag is loaded with koffA, same K order as A

    const _Float16* arow = A  + (size_t)(tm + nm) * K;
    const _Float16* brow = BT + (size_t)(tn + nm) * ldbt;

    v8f acc = {};
    for (int kc = 0; kc < K; kc += 32) {
        if (kc + 128 < K) {
            __builtin_prefetch(arow + kc + 128, 0, 1);   // global_prefetch_b8
            __builtin_prefetch(brow + kc + 128, 0, 1);
        }
        v16h a = cat8(ld8(arow + kc + koffA), ld8(arow + kc + 16 + koffA));
        v16h b = cat8(ld8(brow + kc + koffA), ld8(brow + kc + 16 + koffA));
        acc = __builtin_amdgcn_wmma_f32_16x16x32_f16(
            false, a, false, b, (short)0, acc, false, false);
    }

    float bv = bias ? bias[tn + nm] : 0.f;
#pragma unroll
    for (int r = 0; r < 8; ++r) {
        int m = tm + r + 8 * hi;          // C layout: VGPR r -> M = r + 8*hi
        C[(size_t)m * Nc + tn + nm] = (_Float16)(acc[r] + bv);
    }
}

// ------------------------- fused pair kernel -------------------------------
// One wave per 16-pair tile. Both MLP heads run through WMMA:
//   classifier: A = relu(ca'[i]+cb[j]) [16x512], B = cw2t  [512x16 padded]
//   strength:   A = relu(sa'[i]+sb[j]) [16x256], B = sw2t  [256x16 padded]
// A-fragments built with packed f16 ops; epilogue via LDS.

__global__ __launch_bounds__(128)
void pair_kernel(const _Float16* __restrict__ ca,   // has cb1 folded in
                 const _Float16* __restrict__ cb,
                 const _Float16* __restrict__ cw2t,
                 const float* __restrict__ cb2,
                 const _Float16* __restrict__ sa,   // has sb1 folded in
                 const _Float16* __restrict__ sb,
                 const _Float16* __restrict__ sw2t,
                 const float* __restrict__ sb2,
                 float* __restrict__ out_types,
                 float* __restrict__ out_arg,
                 float* __restrict__ out_str) {
    __shared__ float lds[4][16][17];   // logits tile, pad to dodge bank conflicts
    __shared__ float ldss[4][16];      // strength logit per row

    int lane = threadIdx.x & 31;
    int wave = threadIdx.x >> 5;
    int tile = blockIdx.x * 4 + wave;
    int p0 = tile * 16;
    int m  = lane & 15;
    int hi = lane >> 4;
    int p  = p0 + m;
    int koffA = 8 * hi;
    int koffB = 16 * hi;

    // invert flat upper-triangular index: p -> (i, j), i < j
    double q = (double)(2 * NN - 1);
    int i = (int)((q - sqrt(q * q - 8.0 * (double)p)) * 0.5);
    if (i < 0) i = 0;
    if (i > NN - 2) i = NN - 2;
    while (i < NN - 2 && tri_start(i + 1) <= p) ++i;
    while (i > 0 && tri_start(i) > p) --i;
    int j = p - tri_start(i) + i + 1;

    // ---- classifier head: 16 WMMA chunks over K=512 ----
    {
        const _Float16* cai  = ca + (size_t)i * DD;
        const _Float16* cbj  = cb + (size_t)j * DD;
        const _Float16* brow = cw2t + (size_t)m * DD;

        v8f acc = {};
        for (int kc = 0; kc < DD; kc += 32) {
            if (kc + 128 < DD) {
                __builtin_prefetch(cai + kc + 128, 0, 1);
                __builtin_prefetch(cbj + kc + 128, 0, 1);
            }
            v8h h0 = relu8(ld8(cai + kc + koffA) + ld8(cbj + kc + koffA));
            v8h h1 = relu8(ld8(cai + kc + 16 + koffA) + ld8(cbj + kc + 16 + koffA));
            v16h a = cat8(h0, h1);
            v16h b = cat8(ld8(brow + kc + koffA), ld8(brow + kc + 16 + koffA));
            acc = __builtin_amdgcn_wmma_f32_16x16x32_f16(
                false, a, false, b, (short)0, acc, false, false);
        }
#pragma unroll
        for (int r = 0; r < 8; ++r)
            lds[wave][r + 8 * hi][m] = acc[r];
    }

    // ---- strength head: 8 WMMA chunks over K=256 ----
    {
        const _Float16* sai  = sa + (size_t)i * (DD / 2);
        const _Float16* sbj  = sb + (size_t)j * (DD / 2);
        const _Float16* brow = sw2t + (size_t)m * (DD / 2);

        v8f acc = {};
        for (int kc = 0; kc < DD / 2; kc += 32) {
            v8h h0 = relu8(ld8(sai + kc + koffA) + ld8(sbj + kc + koffA));
            v8h h1 = relu8(ld8(sai + kc + 16 + koffA) + ld8(sbj + kc + 16 + koffA));
            v16h a = cat8(h0, h1);
            v16h b = cat8(ld8(brow + kc + koffA), ld8(brow + kc + 16 + koffA));
            acc = __builtin_amdgcn_wmma_f32_16x16x32_f16(
                false, a, false, b, (short)0, acc, false, false);
        }
        if (m == 0) {                      // column n==0 holds the strength logit
#pragma unroll
            for (int r = 0; r < 8; ++r)
                ldss[wave][r + 8 * hi] = acc[r];
        }
    }

    __syncthreads();

    if (lane < 16) {  // row m == lane; p already computed for this row
        float v[10];
        float mx = -INFINITY;
        int am = 0;
        for (int n = 0; n < 10; ++n) {
            v[n] = lds[wave][lane][n] + cb2[n];
            if (v[n] > mx) { mx = v[n]; am = n; }
        }
        float s = 0.f;
        for (int n = 0; n < 10; ++n) s += expf(v[n] - mx);
        float lse = mx + logf(s);
        for (int n = 0; n < 10; ++n) out_types[(size_t)p * 10 + n] = v[n] - lse;
        out_arg[p] = (float)am;

        float z = ldss[wave][lane] + sb2[0];
        out_str[p] = 1.f / (1.f + expf(-z));
    }
}

// ------------------------- host launcher -----------------------------------

extern "C" void kernel_launch(void* const* d_in, const int* in_sizes, int n_in,
                              void* d_out, int out_size, void* d_ws, size_t ws_size,
                              hipStream_t stream) {
    const float* features = (const float*)d_in[0];
    const float* cw1 = (const float*)d_in[1];
    const float* cb1 = (const float*)d_in[2];
    const float* cw2 = (const float*)d_in[3];
    const float* cb2 = (const float*)d_in[4];
    const float* sw1 = (const float*)d_in[5];
    const float* sb1 = (const float*)d_in[6];
    const float* sw2 = (const float*)d_in[7];
    const float* sb2 = (const float*)d_in[8];

    // workspace layout (bytes), all 16B-aligned; total ~3.52 MB
    char* ws = (char*)d_ws;
    _Float16* fA16 = (_Float16*)(ws);               // 512*512  f16
    _Float16* cw1t = (_Float16*)(ws + 524288);      // 512*1024 f16 (transposed)
    _Float16* sw1t = (_Float16*)(ws + 1572864);     // 256*1024 f16 (transposed)
    _Float16* cw2t = (_Float16*)(ws + 2097152);     // 16*512   f16 (padded)
    _Float16* sw2t = (_Float16*)(ws + 2113536);     // 16*256   f16 (padded)
    _Float16* ca   = (_Float16*)(ws + 2121728);     // 512*512  f16 (+cb1 folded)
    _Float16* cb   = (_Float16*)(ws + 2646016);     // 512*512  f16
    _Float16* sa   = (_Float16*)(ws + 3170304);     // 512*256  f16 (+sb1 folded)
    _Float16* sb   = (_Float16*)(ws + 3432448);     // 512*256  f16

    // 1) precision conversion / weight transposition
    cvt_f32_to_f16<<<1024, 256, 0, stream>>>(features, fA16, NN * DD);
    transpose_to_f16<<<2048, 256, 0, stream>>>(cw1, cw1t, 2 * DD, DD);      // [1024x512]
    transpose_to_f16<<<1024, 256, 0, stream>>>(sw1, sw1t, 2 * DD, DD / 2);  // [1024x256]
    build_cw2t<<<32, 256, 0, stream>>>(cw2, cw2t);
    build_sw2t<<<16, 256, 0, stream>>>(sw2, sw2t);

    // 2) per-node projections via WMMA GEMMs (linear-split trick), biases folded
    proj_gemm_f16<<<256, 128, 0, stream>>>(fA16, cw1t,      2 * DD, ca, cb1,     NN, DD, DD);
    proj_gemm_f16<<<256, 128, 0, stream>>>(fA16, cw1t + DD, 2 * DD, cb, nullptr, NN, DD, DD);
    proj_gemm_f16<<<128, 128, 0, stream>>>(fA16, sw1t,      2 * DD, sa, sb1,     NN, DD, DD / 2);
    proj_gemm_f16<<<128, 128, 0, stream>>>(fA16, sw1t + DD, 2 * DD, sb, nullptr, NN, DD, DD / 2);

    // 3) fused pair kernel: 8176 tiles of 16 pairs, 4 waves/block
    float* out_types = (float*)d_out;
    float* out_arg   = out_types + (size_t)NPAIRS * 10;
    float* out_str   = out_arg + NPAIRS;
    pair_kernel<<<2044, 128, 0, stream>>>(ca, cb, cw2t, cb2,
                                          sa, sb, sw2t, sb2,
                                          out_types, out_arg, out_str);
}
